// Mamba2VisionMixer_89936615178951
// MI455X (gfx1250) — compile-verified, matrix operand order checked by reading, not run on hardware
//
#include <hip/hip_runtime.h>
#include <hip/hip_bf16.h>

// ---------------------------------------------------------------------------
// Mamba2 vision mixer forward, CDNA5 (gfx1250), wave32, WMMA bf16 pipeline.
// WMMA operands staged in LDS pre-swizzled to the ISA 7.12.2 fragment layout
// (contiguous 32B v16bf reads). TDM tensor_load_to_lds used for the linear
// Bc chunk copy in the SSD kernel.
// ---------------------------------------------------------------------------

typedef __attribute__((ext_vector_type(16))) __bf16 v16bf;
typedef __attribute__((ext_vector_type(8)))  float  v8f;
typedef __attribute__((ext_vector_type(4)))  unsigned tdm_u4;
typedef __attribute__((ext_vector_type(8)))  int      tdm_i8;
typedef __attribute__((ext_vector_type(4)))  int      tdm_i4;

#define NTOK   16384      // B*L
#define LSEQ   4096
#define DMODEL 512
#define DINNER 1024
#define NH     16
#define HD     64
#define NCHUNK 64
#define CH     64
#define NPAD   192        // W_xdbl N padded 144 -> 192

__device__ __forceinline__ __bf16 f2bf(float f) {
  union { float f; unsigned u; } v; v.f = f;
  unsigned r = v.u + 0x7FFFu + ((v.u >> 16) & 1u);
  unsigned short h = (unsigned short)(r >> 16);
  __bf16 o; __builtin_memcpy(&o, &h, 2); return o;
}
__device__ __forceinline__ float bf2f(__bf16 b) {
  unsigned short h; __builtin_memcpy(&h, &b, 2);
  union { unsigned u; float f; } v; v.u = ((unsigned)h) << 16; return v.f;
}
__device__ __forceinline__ v8f wmma_bf16(v16bf a, v16bf b, v8f c) {
  return __builtin_amdgcn_wmma_f32_16x16x32_bf16(false, a, false, b, (short)0, c, false, false);
}
__device__ __forceinline__ float siluf(float x) { return x / (1.f + __expf(-x)); }

// ---------------------------------------------------------------------------
__global__ void k_cvt_bf16(const float* __restrict__ in, __bf16* __restrict__ out, long n) {
  long i = (long)blockIdx.x * blockDim.x + threadIdx.x;
  if (i < n) out[i] = f2bf(in[i]);
}
__global__ void k_cvt_pad(const float* __restrict__ in, __bf16* __restrict__ out,
                          int rows, int ci, int co) {
  long i = (long)blockIdx.x * blockDim.x + threadIdx.x;
  if (i >= (long)rows * co) return;
  int c = (int)(i % co); long r = i / co;
  out[i] = (c < ci) ? f2bf(in[r * ci + c]) : f2bf(0.f);
}

// ---------------------------------------------------------------------------
// WMMA GEMM: C[MxN] = A[MxK] * B[KxN], bf16 row-major, M%128==0, N%64==0,
// K%32==0. 128x64 block tile, 256 threads / 8 waves, wave = one 16-row strip
// with 4 accumulators -> A-fragment reused across 4 WMMAs per K step.
// ---------------------------------------------------------------------------
__global__ void k_gemm_bf16(const __bf16* __restrict__ A, const __bf16* __restrict__ B,
                            float* __restrict__ Cf, __bf16* __restrict__ Cb,
                            int M, int N, int K) {
  __shared__ alignas(32) __bf16 AsF[8 * 32 * 16];   // [strip][lane][h]
  __shared__ alignas(32) __bf16 BsF[4 * 32 * 16];   // [nj][lane][h]
  const int tid  = threadIdx.x;
  const int lane = tid & 31;
  const int wave = tid >> 5;
  const int rowBase = blockIdx.y * 128;
  const int colBase = blockIdx.x * 64;

  // B staging coords: 8 contiguous n -> scattered b16 frag stores
  const int br = (tid * 8) >> 6;        // 0..31 k row
  const int bc = (tid * 8) & 63;        // n offset (mult of 8)
  const int bh  = ((br >> 4) & 1) * 8 + (br & 7);
  const int bhi = ((br >> 3) & 1) << 4;

  const v16bf* Av = (const v16bf*)AsF;
  const v16bf* Bv = (const v16bf*)BsF;

  v8f acc[4] = {};
  for (int k0 = 0; k0 < K; k0 += 32) {
    #pragma unroll
    for (int jj = 0; jj < 2; ++jj) {    // A tile 128x32: 2 groups of 8 halves
      int e = tid * 16 + jj * 8;
      int r = e >> 5, c = e & 31;
      uint4 ld = *(const uint4*)(A + (size_t)(rowBase + r) * K + k0 + c);
      int lane_s = (r & 15) | (((c >> 3) & 1) << 4);
      int h0 = ((c >> 4) & 1) * 8;
      *(uint4*)(AsF + (((r >> 4) * 32 + lane_s) * 16 + h0)) = ld;
    }
    { // B tile 32x64
      union { uint4 u; __bf16 b[8]; } pk;
      pk.u = *(const uint4*)(B + (size_t)(k0 + br) * N + colBase + bc);
      #pragma unroll
      for (int j = 0; j < 8; ++j) {
        int n = bc + j;
        BsF[(((n >> 4) * 32 + ((n & 15) | bhi)) * 16) + bh] = pk.b[j];
      }
    }
    if (k0 + 32 < K) {  // prefetch next K tiles (global_prefetch_b8)
      __builtin_prefetch(A + (size_t)(rowBase + (tid >> 1)) * K + k0 + 32 + (tid & 1) * 16, 0, 1);
      __builtin_prefetch(B + (size_t)(k0 + 32 + br) * N + colBase + bc, 0, 1);
    }
    __syncthreads();
    v16bf a = Av[wave * 32 + lane];
    #pragma unroll
    for (int nj = 0; nj < 4; ++nj)
      acc[nj] = wmma_bf16(a, Bv[nj * 32 + lane], acc[nj]);
    __syncthreads();
  }
  const int hi = lane >> 4, cl = lane & 15;
  #pragma unroll
  for (int nj = 0; nj < 4; ++nj) {
    #pragma unroll
    for (int r = 0; r < 8; ++r) {
      size_t row = rowBase + wave * 16 + r + hi * 8;
      size_t col = colBase + nj * 16 + cl;
      if (Cf) Cf[row * N + col] = acc[nj][r];
      else    Cb[row * N + col] = f2bf(acc[nj][r]);
    }
  }
}

// ---------------------------------------------------------------------------
// Depthwise conv(4, SAME pad_lo=1) + SiLU for xi and z branches.
// ---------------------------------------------------------------------------
__global__ void k_conv_silu(const __bf16* __restrict__ xz,
                            const float* __restrict__ kx, const float* __restrict__ bx,
                            const float* __restrict__ kz, const float* __restrict__ bz,
                            float* __restrict__ xif, __bf16* __restrict__ xibf,
                            float* __restrict__ zsf) {
  long i = (long)blockIdx.x * 256 + threadIdx.x;
  if (i >= (long)NTOK * DINNER) return;
  int  ch  = (int)(i & (DINNER - 1));
  long tok = i >> 10;
  int  l   = (int)(tok & (LSEQ - 1));
  long seq0 = tok - l;
  float ax = bx[ch], az = bz[ch];
  #pragma unroll
  for (int w = 0; w < 4; ++w) {
    int t = l + w - 1;
    if (t >= 0 && t < LSEQ) {
      const __bf16* p = xz + ((seq0 + t) << 11);
      ax += bf2f(p[ch])          * kx[w * DINNER + ch];
      az += bf2f(p[DINNER + ch]) * kz[w * DINNER + ch];
    }
  }
  float sx = siluf(ax), sz = siluf(az);
  xif[i] = sx; xibf[i] = f2bf(sx); zsf[i] = sz;
}

// ---------------------------------------------------------------------------
__global__ void k_dt(const float* __restrict__ xdbl, const float* __restrict__ dt_bias,
                     const float* __restrict__ A_log,
                     float* __restrict__ dtg, float* __restrict__ dAg) {
  long i = (long)blockIdx.x * 256 + threadIdx.x;
  if (i >= (long)NTOK * NH) return;
  long tok = i >> 4; int h = (int)(i & 15);
  float v = xdbl[tok * NPAD + h] + dt_bias[h];
  float sp = (v > 20.f) ? v : log1pf(__expf(v));
  dtg[i] = sp;
  dAg[i] = -__expf(A_log[h]) * sp;
}

// Xst[b,c,h,p,s] = bf16( xi[b, c*64+s, h*64+p] * dt[b, c*64+s, h] )
__global__ void k_xscale(const float* __restrict__ xif, const float* __restrict__ dtg,
                         __bf16* __restrict__ Xst) {
  long o = (long)blockIdx.x * 256 + threadIdx.x;
  if (o >= (long)NTOK * DINNER) return;
  int s = (int)(o & 63); long r = o >> 6;
  int p = (int)(r & 63); r >>= 6;
  int h = (int)(r & 15); r >>= 4;
  int c = (int)(r & 63); int b = (int)(r >> 6);
  long tok = (long)b * LSEQ + c * CH + s;
  Xst[o] = f2bf(xif[tok * DINNER + h * HD + p] * dtg[tok * NH + h]);
}

// split xdbl (stride 192) -> B (cols 16..79) / C (cols 80..143), bf16
__global__ void k_split_bc(const float* __restrict__ xdbl,
                           __bf16* __restrict__ Bcbf, __bf16* __restrict__ Ccbf) {
  long i = (long)blockIdx.x * 256 + threadIdx.x;
  if (i >= (long)NTOK * 128) return;
  long tok = i >> 7; int n = (int)(i & 127);
  if (n < 64) Bcbf[tok * 64 + n]        = f2bf(xdbl[tok * NPAD + 16 + n]);
  else        Ccbf[tok * 64 + (n - 64)] = f2bf(xdbl[tok * NPAD + 80 + (n - 64)]);
}

// ---------------------------------------------------------------------------
__global__ void k_scan(const float* __restrict__ dAg, float* __restrict__ Acum) {
  __shared__ float s[CH];
  int idx = blockIdx.x;
  int c = idx & 63, h = (idx >> 6) & 15, b = idx >> 10;
  int t = threadIdx.x;
  long tok = (long)b * LSEQ + c * CH + t;
  s[t] = dAg[tok * NH + h];
  __syncthreads();
  if (t == 0) {
    float acc = 0.f;
    for (int j = 0; j < CH; ++j) { acc += s[j]; s[j] = acc; }
  }
  __syncthreads();
  Acum[(((long)(b * NH + h)) * NCHUNK + c) * CH + t] = s[t];
}

// ---------------------------------------------------------------------------
// SSD intra-chunk, one block (256 thr) per (b,c,h).
//   G  = C * B^T (decay-masked)  ->  Yd = G' * X  ->  S = Bd^T * X
// Bc chunk (contiguous 8KB) is fetched by the Tensor Data Mover into LDS
// (overlapped with the swizzled manual staging), consumed by the Bd^T build.
// ---------------------------------------------------------------------------
__global__ void k_ssd_chunk(const __bf16* __restrict__ Ccbf, const __bf16* __restrict__ Bcbf,
                            const __bf16* __restrict__ Xst, const float* __restrict__ Acum,
                            float* __restrict__ Yd, float* __restrict__ states) {
  __shared__ alignas(32) __bf16 CsA[2 * 4 * 32 * 16];  // C as A-op   (M=l,K=n)
  __shared__ alignas(32) __bf16 BsT[2 * 4 * 32 * 16];  // B^T as B-op (K=n,N=s)
  __shared__ alignas(32) __bf16 XsB[2 * 4 * 32 * 16];  // X as B-op   (K=s,N=p)
  __shared__ alignas(32) __bf16 GsA[2 * 4 * 32 * 16];  // G / Bd^T as A-op
  __shared__ alignas(16) __bf16 BcLin[64 * 64];        // linear Bc chunk (TDM dest)
  __shared__ float  ac[64];
  int idx = blockIdx.x;
  int c = idx & 63, h = (idx >> 6) & 15, b = idx >> 10;
  int tid = threadIdx.x, lane = tid & 31, wave = tid >> 5;
  long tokBase = (long)b * LSEQ + c * CH;
  long xstBase = (((long)((b * NCHUNK + c) * NH + h))) * 4096;

#if __has_builtin(__builtin_amdgcn_tensor_load_to_lds) && __has_builtin(__builtin_amdgcn_s_wait_tensorcnt)
  if (wave == 0) {
    // D# for a 64x64 2-byte tile, fully linear copy into BcLin.
    unsigned long long ga = (unsigned long long)(size_t)(Bcbf + tokBase * 64);
    unsigned lds = (unsigned)(size_t)(&BcLin[0]);
    tdm_u4 g0;
    g0[0] = 1u;                                        // count=1 user descriptor
    g0[1] = lds;                                       // lds_addr
    g0[2] = (unsigned)(ga & 0xFFFFFFFFu);              // global_addr[31:0]
    g0[3] = (unsigned)((ga >> 32) & 0x01FFFFFFu) | (2u << 30);  // [56:32] | type=2
    tdm_i8 g1;
    g1[0] = 0x00010000;          // data_size=1 (2 bytes), no multicast/pad
    g1[1] = (64 << 16);          // tensor_dim0[15:0]=64
    g1[2] = (64 << 16);          // tensor_dim0[31:16]=0 | tensor_dim1[15:0]=64
    g1[3] = (64 << 16);          // tensor_dim1[31:16]=0 | tile_dim0=64
    g1[4] = 64;                  // tile_dim1=64, tile_dim2=0
    g1[5] = 64;                  // tensor_dim0_stride[31:0]=64
    g1[6] = 0; g1[7] = 0;
    tdm_i4 gz4 = {0, 0, 0, 0};
    tdm_i8 gz8 = {0, 0, 0, 0, 0, 0, 0, 0};
    __builtin_amdgcn_tensor_load_to_lds(g0, g1, gz4, gz4, gz8, 0);
  }
#else
  #pragma unroll
  for (int jj = 0; jj < 2; ++jj) {
    int e = tid + jj * 256;
    *(uint4*)(BcLin + e * 8) = *(const uint4*)(Bcbf + tokBase * 64 + e * 8);
  }
#endif

  if (tid < 64) ac[tid] = Acum[(((long)(b * NH + h)) * NCHUNK + c) * CH + tid];

  // stage C (A-frag), B^T (B-frag), X (B-frag): contiguous b128 paths
  #pragma unroll
  for (int jj = 0; jj < 2; ++jj) {
    int g = tid + jj * 256;               // 512 groups of 8 halves
    int row = g >> 3, c8 = (g & 7) * 8;   // row = l|s|p ; c8 = k offset
    int kk = c8 >> 5, h0 = ((c8 >> 4) & 1) * 8, khi = ((c8 >> 3) & 1) << 4;
    { // C: element (m=row(l), k=c8..(n))
      uint4 ld = *(const uint4*)(Ccbf + (tokBase + row) * 64 + c8);
      *(uint4*)(CsA + (((kk * 4 + (row >> 4)) * 32 + ((row & 15) | khi)) * 16 + h0)) = ld;
    }
    { // B^T: element (k=c8..(n), n=row(s)) from Bc[s][n]
      uint4 ld = *(const uint4*)(Bcbf + (tokBase + row) * 64 + c8);
      *(uint4*)(BsT + (((kk * 4 + (row >> 4)) * 32 + ((row & 15) | khi)) * 16 + h0)) = ld;
    }
    { // X: element (k=c8..(s), n=row(p)) from Xst[p][s]
      uint4 ld = *(const uint4*)(Xst + xstBase + row * 64 + c8);
      *(uint4*)(XsB + (((kk * 4 + (row >> 4)) * 32 + ((row & 15) | khi)) * 16 + h0)) = ld;
    }
  }
#if __has_builtin(__builtin_amdgcn_tensor_load_to_lds) && __has_builtin(__builtin_amdgcn_s_wait_tensorcnt)
  if (wave == 0) __builtin_amdgcn_s_wait_tensorcnt(0);
#endif
  __syncthreads();
  const float aend = ac[63];
  const v16bf* Cv = (const v16bf*)CsA;
  const v16bf* Tv = (const v16bf*)BsT;
  const v16bf* Xv = (const v16bf*)XsB;
  const v16bf* Gv = (const v16bf*)GsA;
  const int mi = wave >> 1, nj0 = (wave & 1) * 2;
  const int hi = lane >> 4, cl = lane & 15;

  // ---- G = C * B^T, causal decay at store, scatter into A-frag layout ----
  #pragma unroll
  for (int t = 0; t < 2; ++t) {
    int nj = nj0 + t;
    v8f acc = {};
    #pragma unroll
    for (int kk = 0; kk < 2; ++kk)
      acc = wmma_bf16(Cv[(kk * 4 + mi) * 32 + lane], Tv[(kk * 4 + nj) * 32 + lane], acc);
    int glane = ((lane >> 3) & 1) << 4;
    int gh    = (nj & 1) * 8 + (lane & 7);
    int gkk   = nj >> 1;
    #pragma unroll
    for (int r = 0; r < 8; ++r) {
      int l = mi * 16 + r + hi * 8, s = nj * 16 + cl;
      float v = (s <= l) ? acc[r] * __expf(ac[l] - ac[s]) : 0.f;
      GsA[((gkk * 4 + mi) * 32 + ((l & 15) | glane)) * 16 + gh] = f2bf(v);
    }
  }
  __syncthreads();

  // ---- Y_diag = G' * X ----
  #pragma unroll
  for (int t = 0; t < 2; ++t) {
    int nj = nj0 + t;
    v8f acc = {};
    #pragma unroll
    for (int kk = 0; kk < 2; ++kk)
      acc = wmma_bf16(Gv[(kk * 4 + mi) * 32 + lane], Xv[(kk * 4 + nj) * 32 + lane], acc);
    #pragma unroll
    for (int r = 0; r < 8; ++r) {
      int l = mi * 16 + r + hi * 8, p = nj * 16 + cl;
      Yd[(tokBase + l) * DINNER + h * HD + p] = acc[r];
    }
  }
  __syncthreads();

  // ---- Bd^T = (B .* exp(aend-Acum[s]))^T into GsA (A-frag, M=n, K=s) ----
  #pragma unroll
  for (int jj = 0; jj < 2; ++jj) {
    int g = tid + jj * 256;
    int n = g & 63, s0 = (g >> 6) * 8;
    union { uint4 u; __bf16 b[8]; } pk;
    #pragma unroll
    for (int j = 0; j < 8; ++j)
      pk.b[j] = f2bf(bf2f(BcLin[(s0 + j) * 64 + n]) * __expf(aend - ac[s0 + j]));
    int kk = s0 >> 5, h0 = ((s0 >> 4) & 1) * 8, khi = ((s0 >> 3) & 1) << 4;
    *(uint4*)(GsA + (((kk * 4 + (n >> 4)) * 32 + ((n & 15) | khi)) * 16 + h0)) = pk.u;
  }
  __syncthreads();

  // ---- S[n,p] = Bd^T * X   (stored transposed: states[..., p, n]) ----
  long sBase = xstBase;
  #pragma unroll
  for (int t = 0; t < 2; ++t) {
    int nj = nj0 + t;
    v8f acc = {};
    #pragma unroll
    for (int kk = 0; kk < 2; ++kk)
      acc = wmma_bf16(Gv[(kk * 4 + mi) * 32 + lane], Xv[(kk * 4 + nj) * 32 + lane], acc);
    #pragma unroll
    for (int r = 0; r < 8; ++r) {
      int n = mi * 16 + r + hi * 8, p = nj * 16 + cl;
      states[sBase + p * 64 + n] = acc[r];
    }
  }
}

// ---------------------------------------------------------------------------
__global__ void k_chunk_scan(const float* __restrict__ states, const float* __restrict__ Acum,
                             float* __restrict__ prevstates) {
  int h = blockIdx.x & 15, b = blockIdx.x >> 4;
  int tid = threadIdx.x;
  float carry[16];
  #pragma unroll
  for (int j = 0; j < 16; ++j) carry[j] = 0.f;
  for (int c = 0; c < NCHUNK; ++c) {
    long base = (((long)((b * NCHUNK + c) * NH + h))) * 4096;
    float f = __expf(Acum[(((long)(b * NH + h)) * NCHUNK + c) * CH + 63]);
    #pragma unroll
    for (int j = 0; j < 16; ++j) {
      long e = base + j * 256 + tid;
      prevstates[e] = carry[j];
      carry[j] = carry[j] * f + states[e];
    }
  }
}

// ---------------------------------------------------------------------------
// Y_off = exp(Acum[l]) * (C @ S_prev);  Y = Yd + Y_off + D[h]*xh  (in place)
// ---------------------------------------------------------------------------
__global__ void k_ssd_off(const __bf16* __restrict__ Ccbf, const float* __restrict__ prevst,
                          const float* __restrict__ Acum, const float* __restrict__ xif,
                          const float* __restrict__ Dv, float* __restrict__ Y) {
  __shared__ alignas(32) __bf16 CsA[2 * 4 * 32 * 16];
  __shared__ alignas(32) __bf16 SsB[2 * 4 * 32 * 16];
  __shared__ float  ac[64];
  int idx = blockIdx.x;
  int c = idx & 63, h = (idx >> 6) & 15, b = idx >> 10;
  int tid = threadIdx.x, lane = tid & 31, wave = tid >> 5;
  long tokBase = (long)b * LSEQ + c * CH;
  long sBase = (((long)((b * NCHUNK + c) * NH + h))) * 4096;
  if (tid < 64) ac[tid] = Acum[(((long)(b * NH + h)) * NCHUNK + c) * CH + tid];
  #pragma unroll
  for (int jj = 0; jj < 2; ++jj) {
    int g = tid + jj * 256;
    int row = g >> 3, c8 = (g & 7) * 8;
    int kk = c8 >> 5, h0 = ((c8 >> 4) & 1) * 8, khi = ((c8 >> 3) & 1) << 4;
    { // C as A-op
      uint4 ld = *(const uint4*)(Ccbf + (tokBase + row) * 64 + c8);
      *(uint4*)(CsA + (((kk * 4 + (row >> 4)) * 32 + ((row & 15) | khi)) * 16 + h0)) = ld;
    }
    { // S as B-op from prevT[p][n], f32->bf16
      const float* sp = prevst + sBase + row * 64 + c8;
      float4 f0 = *(const float4*)(sp);
      float4 f1 = *(const float4*)(sp + 4);
      union { uint4 u; __bf16 b[8]; } pk;
      pk.b[0] = f2bf(f0.x); pk.b[1] = f2bf(f0.y); pk.b[2] = f2bf(f0.z); pk.b[3] = f2bf(f0.w);
      pk.b[4] = f2bf(f1.x); pk.b[5] = f2bf(f1.y); pk.b[6] = f2bf(f1.z); pk.b[7] = f2bf(f1.w);
      *(uint4*)(SsB + (((kk * 4 + (row >> 4)) * 32 + ((row & 15) | khi)) * 16 + h0)) = pk.u;
    }
  }
  __syncthreads();
  const v16bf* Cv = (const v16bf*)CsA;
  const v16bf* Sv = (const v16bf*)SsB;
  const int mi = wave >> 1, nj0 = (wave & 1) * 2;
  const int hi = lane >> 4, cl = lane & 15;
  const float dh = Dv[h];
  #pragma unroll
  for (int t = 0; t < 2; ++t) {
    int nj = nj0 + t;
    v8f acc = {};
    #pragma unroll
    for (int kk = 0; kk < 2; ++kk)
      acc = wmma_bf16(Cv[(kk * 4 + mi) * 32 + lane], Sv[(kk * 4 + nj) * 32 + lane], acc);
    #pragma unroll
    for (int r = 0; r < 8; ++r) {
      int l = mi * 16 + r + hi * 8, p = nj * 16 + cl;
      long o = (tokBase + l) * DINNER + h * HD + p;
      Y[o] = Y[o] + acc[r] * __expf(ac[l]) + dh * xif[o];
    }
  }
}

// ---------------------------------------------------------------------------
__global__ void k_gate_rms(const float* __restrict__ Y, const float* __restrict__ zs,
                           const float* __restrict__ nw, __bf16* __restrict__ out) {
  __shared__ float red[256];
  long tok = blockIdx.x;
  int tid = threadIdx.x;
  float yv[4]; float ss = 0.f;
  #pragma unroll
  for (int j = 0; j < 4; ++j) {
    int d = tid + j * 256;
    float v = Y[tok * DINNER + d] * zs[tok * DINNER + d];
    yv[j] = v; ss += v * v;
  }
  red[tid] = ss; __syncthreads();
  for (int s = 128; s > 0; s >>= 1) {
    if (tid < s) red[tid] += red[tid + s];
    __syncthreads();
  }
  float scale = rsqrtf(red[0] / (float)DINNER + 1e-5f);
  #pragma unroll
  for (int j = 0; j < 4; ++j) {
    int d = tid + j * 256;
    out[tok * DINNER + d] = f2bf(yv[j] * scale * nw[d]);
  }
}

// ---------------------------------------------------------------------------
extern "C" void kernel_launch(void* const* d_in, const int* in_sizes, int n_in,
                              void* d_out, int out_size, void* d_ws, size_t ws_size,
                              hipStream_t stream) {
  const float* x       = (const float*)d_in[0];
  const float* W_in    = (const float*)d_in[1];
  const float* convxk  = (const float*)d_in[2];
  const float* convxb  = (const float*)d_in[3];
  const float* convzk  = (const float*)d_in[4];
  const float* convzb  = (const float*)d_in[5];
  const float* W_xdbl  = (const float*)d_in[6];
  const float* dt_bias = (const float*)d_in[7];
  const float* A_log   = (const float*)d_in[8];
  const float* Dv      = (const float*)d_in[9];
  const float* norm_w  = (const float*)d_in[10];
  const float* W_out   = (const float*)d_in[11];
  float* outp = (float*)d_out;

  size_t off = 0;
  char* base = (char*)d_ws;
  auto alloc = [&](size_t bytes) -> void* {
    off = (off + 255) & ~(size_t)255;
    void* p = base + off; off += bytes; return p;
  };
  __bf16* xbf    = (__bf16*)alloc((size_t)NTOK * DMODEL * 2);
  __bf16* Winbf  = (__bf16*)alloc((size_t)DMODEL * 2048 * 2);
  __bf16* Wxbf   = (__bf16*)alloc((size_t)DINNER * NPAD * 2);
  __bf16* Wobf   = (__bf16*)alloc((size_t)DINNER * DMODEL * 2);
  __bf16* xzbf   = (__bf16*)alloc((size_t)NTOK * 2048 * 2);
  float*  xif    = (float*) alloc((size_t)NTOK * DINNER * 4);
  __bf16* xibf   = (__bf16*)alloc((size_t)NTOK * DINNER * 2);
  float*  zsf    = (float*) alloc((size_t)NTOK * DINNER * 4);
  float*  xdbl   = (float*) alloc((size_t)NTOK * NPAD * 4);
  float*  dtg    = (float*) alloc((size_t)NTOK * NH * 4);
  float*  dAg    = (float*) alloc((size_t)NTOK * NH * 4);
  __bf16* Xst    = (__bf16*)alloc((size_t)NTOK * DINNER * 2);
  __bf16* Bcbf   = (__bf16*)alloc((size_t)NTOK * 64 * 2);
  __bf16* Ccbf   = (__bf16*)alloc((size_t)NTOK * 64 * 2);
  float*  Acum   = (float*) alloc((size_t)4 * NH * NCHUNK * CH * 4);
  float*  states = (float*) alloc((size_t)4 * NCHUNK * NH * 4096 * 4);
  float*  prevst = (float*) alloc((size_t)4 * NCHUNK * NH * 4096 * 4);
  float*  Ybuf   = (float*) alloc((size_t)NTOK * DINNER * 4);
  __bf16* ynbf   = (__bf16*)alloc((size_t)NTOK * DINNER * 2);
  (void)ws_size; (void)in_sizes; (void)n_in; (void)out_size;

  auto cdiv = [](long a, long b) { return (unsigned)((a + b - 1) / b); };

  // 0) conversions
  k_cvt_bf16<<<cdiv((long)NTOK * DMODEL, 256), 256, 0, stream>>>(x, xbf, (long)NTOK * DMODEL);
  k_cvt_bf16<<<cdiv((long)DMODEL * 2048, 256), 256, 0, stream>>>(W_in, Winbf, (long)DMODEL * 2048);
  k_cvt_pad<<<cdiv((long)DINNER * NPAD, 256), 256, 0, stream>>>(W_xdbl, Wxbf, DINNER, 144, NPAD);
  k_cvt_bf16<<<cdiv((long)DINNER * DMODEL, 256), 256, 0, stream>>>(W_out, Wobf, (long)DINNER * DMODEL);

  // 1) xz = x @ W_in
  k_gemm_bf16<<<dim3(2048 / 64, NTOK / 128), 256, 0, stream>>>(xbf, Winbf, nullptr, xzbf,
                                                               NTOK, 2048, DMODEL);
  // 2) depthwise conv + SiLU
  k_conv_silu<<<cdiv((long)NTOK * DINNER, 256), 256, 0, stream>>>(xzbf, convxk, convxb,
                                                                  convzk, convzb, xif, xibf, zsf);
  // 3) x_dbl = xi @ W_xdbl (padded N=192)
  k_gemm_bf16<<<dim3(NPAD / 64, NTOK / 128), 256, 0, stream>>>(xibf, Wxbf, xdbl, nullptr,
                                                               NTOK, NPAD, DINNER);
  // 4) dt / dA / scaled-transposed X / B,C split
  k_dt<<<cdiv((long)NTOK * NH, 256), 256, 0, stream>>>(xdbl, dt_bias, A_log, dtg, dAg);
  k_xscale<<<cdiv((long)NTOK * DINNER, 256), 256, 0, stream>>>(xif, dtg, Xst);
  k_split_bc<<<cdiv((long)NTOK * 128, 256), 256, 0, stream>>>(xdbl, Bcbf, Ccbf);

  // 5) per-chunk cumsum
  k_scan<<<4 * NH * NCHUNK, CH, 0, stream>>>(dAg, Acum);

  // 6) SSD intra-chunk (3 WMMA phases per block, TDM copy of Bc)
  k_ssd_chunk<<<4 * NH * NCHUNK, 256, 0, stream>>>(Ccbf, Bcbf, Xst, Acum, Ybuf, states);

  // 7) inter-chunk recurrence
  k_chunk_scan<<<4 * NH, 256, 0, stream>>>(states, Acum, prevst);

  // 8) Y_off + D*x combine (WMMA)
  k_ssd_off<<<4 * NH * NCHUNK, 256, 0, stream>>>(Ccbf, prevst, Acum, xif, Dv, Ybuf);

  // 9) gate + RMS norm -> bf16
  k_gate_rms<<<NTOK, 256, 0, stream>>>(Ybuf, zsf, norm_w, ynbf);

  // 10) out = ynorm @ W_out
  k_gemm_bf16<<<dim3(DMODEL / 64, NTOK / 128), 256, 0, stream>>>(ynbf, Wobf, outp, nullptr,
                                                                 NTOK, DMODEL, DINNER);
}
